// RotaryMultiheadAttention_27702539059870
// MI455X (gfx1250) — matrix operand order repaired and checked by reading, not compiled, not run on hardware
//
#include <hip/hip_runtime.h>

typedef _Float16 v16h __attribute__((ext_vector_type(16)));
typedef _Float16 v4h  __attribute__((ext_vector_type(4)));
typedef float    v8f  __attribute__((ext_vector_type(8)));

constexpr int B  = 2;
constexpr int T  = 2048;
constexpr int E  = 1024;
constexpr int H  = 16;
constexpr int HD = 64;
constexpr int E3 = 3 * E;
constexpr int BT = B * T;                 // 4096
constexpr float SCALE = 0.125f;           // 1/sqrt(64)
constexpr float INV_H = 1.0f / 16.0f;

union FragH { v16h h; unsigned u[8]; };

static __device__ __forceinline__ v8f wmma_f16(v16h a, v16h b, v8f c) {
  // 8 args: (neg_a, A, neg_b, B, c_mod, C, reuse_a, reuse_b)
  return __builtin_amdgcn_wmma_f32_16x16x32_f16(false, a, false, b, (short)0, c,
                                                false, false);
}

// Async copy 16 bytes global -> LDS (per lane), tracked by ASYNCcnt.
// lds_off = LDS byte offset (low 32 bits of generic shared pointer),
// must be 16-byte aligned for B128.
static __device__ __forceinline__ void async_copy_b128(unsigned lds_off,
                                                       const void* gptr) {
  asm volatile("global_load_async_to_lds_b128 %0, %1, off"
               :
               : "v"(lds_off), "v"(gptr)
               : "memory");
}

static __device__ __forceinline__ void wait_async0() {
  asm volatile("s_wait_asynccnt 0" ::: "memory");
}

// ---------------------------------------------------------------------------
// zero-fill (a_avg accumulator region of d_out)
// ---------------------------------------------------------------------------
__global__ __launch_bounds__(256) void zero_f32(float* __restrict__ p, int n4) {
  int i = blockIdx.x * 256 + threadIdx.x;
  if (i < n4) {
    float4 z; z.x = 0.f; z.y = 0.f; z.z = 0.f; z.w = 0.f;
    ((float4*)p)[i] = z;
  }
}

// ---------------------------------------------------------------------------
// C[M][N] = A[M][K] * W[N][K]^T + bias[N]   (NT GEMM, f32 out)
// block = 128 threads (4 waves), tile 64x64, K chunks of 32 staged in LDS f16
// ---------------------------------------------------------------------------
template <bool A_HALF>
__global__ __launch_bounds__(128)
void gemm_nt_bias(const void* __restrict__ Ap, const float* __restrict__ W,
                  const float* __restrict__ bias, float* __restrict__ out,
                  int M, int N, int K) {
  __shared__ _Float16 sA[64][34];
  __shared__ _Float16 sB[64][34];

  const int tid  = threadIdx.x;
  const int wave = tid >> 5;
  const int lane = tid & 31;
  const int l16  = lane & 15;
  const int hs   = lane >> 4;          // 0 | 1 : which 16-lane half
  const int m0   = blockIdx.y * 64;
  const int n0   = blockIdx.x * 64;

  const int r0 = tid >> 3;             // 0..15
  const int c4 = (tid & 7) * 4;        // 0..28 step 4

  v8f acc[4] = {};

  for (int k0 = 0; k0 < K; k0 += 32) {
    // prefetch next K chunk while this one is staged/consumed
    if (k0 + 32 < K) {
      if (A_HALF) {
        __builtin_prefetch((const _Float16*)Ap +
                               (size_t)(m0 + r0) * K + k0 + 32 + c4, 0, 0);
      } else {
        __builtin_prefetch((const float*)Ap +
                               (size_t)(m0 + r0) * K + k0 + 32 + c4, 0, 0);
      }
      __builtin_prefetch(W + (size_t)(n0 + r0) * K + k0 + 32 + c4, 0, 0);
    }
#pragma unroll
    for (int i = 0; i < 4; ++i) {
      const int r = r0 + i * 16;
      if (A_HALF) {
        const _Float16* A = (const _Float16*)Ap;
        v4h v = *(const v4h*)(A + (size_t)(m0 + r) * K + k0 + c4);
        sA[r][c4 + 0] = v[0]; sA[r][c4 + 1] = v[1];
        sA[r][c4 + 2] = v[2]; sA[r][c4 + 3] = v[3];
      } else {
        const float* A = (const float*)Ap;
        float4 v = *(const float4*)(A + (size_t)(m0 + r) * K + k0 + c4);
        sA[r][c4 + 0] = (_Float16)v.x; sA[r][c4 + 1] = (_Float16)v.y;
        sA[r][c4 + 2] = (_Float16)v.z; sA[r][c4 + 3] = (_Float16)v.w;
      }
      float4 wv = *(const float4*)(W + (size_t)(n0 + r) * K + k0 + c4);
      sB[r][c4 + 0] = (_Float16)wv.x; sB[r][c4 + 1] = (_Float16)wv.y;
      sB[r][c4 + 2] = (_Float16)wv.z; sB[r][c4 + 3] = (_Float16)wv.w;
    }
    __syncthreads();

    FragH a;
#pragma unroll
    for (int v = 0; v < 8; ++v) {
      const int kk = ((v & 3) * 2) + ((v >= 4) ? 16 : 0) + hs * 8;
      a.u[v] = *(const unsigned*)&sA[wave * 16 + l16][kk];
    }
#pragma unroll
    for (int ns = 0; ns < 4; ++ns) {
      FragH bfr;
#pragma unroll
      for (int v = 0; v < 8; ++v)
        bfr.u[v] = *(const unsigned*)&sB[ns * 16 + l16][((v & 3) * 2) + ((v >= 4) ? 16 : 0) + hs * 8];
      acc[ns] = wmma_f16(a.h, bfr.h, acc[ns]);
    }
    __syncthreads();
  }

#pragma unroll
  for (int ns = 0; ns < 4; ++ns) {
    const int col = n0 + ns * 16 + l16;
    const float bv = bias[col];
#pragma unroll
    for (int i = 0; i < 8; ++i) {
      const int row = m0 + wave * 16 + i + hs * 8;
      out[(size_t)row * N + col] = acc[ns][i] + bv;
    }
  }
}

// ---------------------------------------------------------------------------
// split qkv -> q/v (copy) and k (rotary complex multiply), f32 -> f16,
// layout [B][H][T][HD]
// ---------------------------------------------------------------------------
__global__ __launch_bounds__(256)
void rotary_scatter(const float* __restrict__ qkv, const float* __restrict__ mre,
                    const float* __restrict__ mim, _Float16* __restrict__ qh,
                    _Float16* __restrict__ kh, _Float16* __restrict__ vh) {
  const int PP = E3 / 2;  // pairs per (b,t)
  int idx = blockIdx.x * 256 + threadIdx.x;
  if (idx >= B * T * PP) return;
  const int p = idx % PP;
  const int t = (idx / PP) % T;
  const int b = idx / (PP * T);
  const int c   = p * 2;
  const int sec = c / E;
  const int ce  = c % E;
  const int h   = ce / HD;
  const int d   = ce % HD;
  const size_t src = (size_t)(b * T + t) * E3 + c;
  const float x0 = qkv[src];
  const float x1 = qkv[src + 1];
  const size_t di = ((size_t)((b * H + h) * T + t)) * HD + d;
  if (sec == 0) {
    qh[di] = (_Float16)x0; qh[di + 1] = (_Float16)x1;
  } else if (sec == 1) {
    const size_t mi_idx = (size_t)(b * T + t) * (HD / 2) + (d >> 1);
    const float mr = mre[mi_idx];
    const float mi = mim[mi_idx];
    kh[di]     = (_Float16)(x0 * mr - x1 * mi);
    kh[di + 1] = (_Float16)(x0 * mi + x1 * mr);
  } else {
    vh[di] = (_Float16)x0; vh[di + 1] = (_Float16)x1;
  }
}

// ---------------------------------------------------------------------------
// attention: one block per (b, h, 64-row q tile); 4 waves, 16 q rows each.
// two-pass softmax; P -> a_avg via f32 atomics; O = P*V via WMMA; ctx f16.
// K tile staged with async-to-LDS (ASYNCcnt), V tile staged transposed.
// ---------------------------------------------------------------------------
__global__ __launch_bounds__(128)
void attention(const _Float16* __restrict__ qh, const _Float16* __restrict__ kh,
               const _Float16* __restrict__ vh, const float* __restrict__ pad,
               float* __restrict__ a_avg, _Float16* __restrict__ ctx) {
  // K tile rows padded to 72 halfs = 144 B (16B multiple for async B128; the
  // 36-dword row stride maps the 16-lane fragment gather onto 16 banks).
  __shared__ __align__(16) _Float16 kt[64][72];  // [key][dim]
  __shared__ _Float16 vt[64][66];                // V transposed [dim][key]
  __shared__ _Float16 pbuf[4][16][66];           // per-wave P strip
  __shared__ float    pm[64];

  const int tid  = threadIdx.x;
  const int wave = tid >> 5;
  const int lane = tid & 31;
  const int l16  = lane & 15;
  const int hs   = lane >> 4;

  const int qt = blockIdx.x & 31;
  const int h  = (blockIdx.x >> 5) & 15;
  const int b  = blockIdx.x >> 9;

  const size_t headBase = (size_t)(b * H + h) * T * HD;
  const int qr0 = qt * 64 + wave * 16;

  // Q A-fragments: rows qr0..qr0+15, K (head dim) 0..63 -> two 16x32 frags
  FragH aq[2];
#pragma unroll
  for (int f = 0; f < 2; ++f)
#pragma unroll
    for (int v = 0; v < 8; ++v) {
      const int kk = 32 * f + ((v & 3) * 2) + ((v >= 4) ? 16 : 0) + hs * 8;
      aq[f].u[v] =
          *(const unsigned*)(qh + headBase + (size_t)(qr0 + l16) * HD + kk);
    }

  float m[8], l[8];
#pragma unroll
  for (int i = 0; i < 8; ++i) { m[i] = -3.0e38f; l[i] = 0.0f; }

  const int stg_r = tid >> 1;          // key row 0..63
  const int stg_c = (tid & 1) * 32;    // dim half

  // ------------------ pass 1: running row max / exp-sum ------------------
  for (int kti = 0; kti < T / 64; ++kti) {
    const int key0 = kti * 64;
    const _Float16* kb = kh + headBase + (size_t)key0 * HD;
#pragma unroll
    for (int j = 0; j < 4; ++j) {
      const unsigned loff =
          (unsigned)(uintptr_t)&kt[stg_r][stg_c + j * 8];
      async_copy_b128(loff, kb + (size_t)stg_r * HD + stg_c + j * 8);
    }
    if (tid < 64) pm[tid] = pad[b * T + key0 + tid];
    wait_async0();
    __syncthreads();

#pragma unroll
    for (int ns = 0; ns < 4; ++ns) {
      v8f s = {};
#pragma unroll
      for (int f = 0; f < 2; ++f) {
        FragH bk;
#pragma unroll
        for (int v = 0; v < 8; ++v)
          bk.u[v] =
              *(const unsigned*)&kt[ns * 16 + l16][32 * f + ((v & 3) * 2) + ((v >= 4) ? 16 : 0) + hs * 8];
        s = wmma_f16(aq[f].h, bk.h, s);
      }
      const float pmv = pm[ns * 16 + l16];
#pragma unroll
      for (int i = 0; i < 8; ++i) {
        const float val = s[i] * SCALE + pmv;
        float tm = val;
#pragma unroll
        for (int msk = 8; msk >= 1; msk >>= 1)
          tm = fmaxf(tm, __shfl_xor(tm, msk, 32));
        const float nm = fmaxf(m[i], tm);
        float e = __expf(val - nm);
#pragma unroll
        for (int msk = 8; msk >= 1; msk >>= 1) e += __shfl_xor(e, msk, 32);
        l[i] = l[i] * __expf(m[i] - nm) + e;
        m[i] = nm;
      }
    }
    __syncthreads();
  }

  float rl[8];
#pragma unroll
  for (int i = 0; i < 8; ++i) rl[i] = 1.0f / l[i];

  // ------------------ pass 2: P, a_avg atomics, O = P*V ------------------
  v8f o0 = {}, o1 = {}, o2 = {}, o3 = {};
  v8f* oacc[4] = {&o0, &o1, &o2, &o3};
  union U4H8 { uint4 u; _Float16 hh[8]; };

  for (int kti = 0; kti < T / 64; ++kti) {
    const int key0 = kti * 64;
    const _Float16* kb = kh + headBase + (size_t)key0 * HD;
    const _Float16* vb = vh + headBase + (size_t)key0 * HD;
#pragma unroll
    for (int j = 0; j < 4; ++j) {
      const unsigned loff =
          (unsigned)(uintptr_t)&kt[stg_r][stg_c + j * 8];
      async_copy_b128(loff, kb + (size_t)stg_r * HD + stg_c + j * 8);
      U4H8 uv;
      uv.u = *(const uint4*)(vb + (size_t)stg_r * HD + stg_c + j * 8);
#pragma unroll
      for (int e = 0; e < 8; ++e) vt[stg_c + j * 8 + e][stg_r] = uv.hh[e];
    }
    if (tid < 64) pm[tid] = pad[b * T + key0 + tid];
    wait_async0();
    __syncthreads();

#pragma unroll
    for (int ns = 0; ns < 4; ++ns) {
      v8f s = {};
#pragma unroll
      for (int f = 0; f < 2; ++f) {
        FragH bk;
#pragma unroll
        for (int v = 0; v < 8; ++v)
          bk.u[v] =
              *(const unsigned*)&kt[ns * 16 + l16][32 * f + ((v & 3) * 2) + ((v >= 4) ? 16 : 0) + hs * 8];
        s = wmma_f16(aq[f].h, bk.h, s);
      }
      const float pmv = pm[ns * 16 + l16];
#pragma unroll
      for (int i = 0; i < 8; ++i) {
        const float val = s[i] * SCALE + pmv;
        const float p = __expf(val - m[i]) * rl[i];
        pbuf[wave][i + hs * 8][ns * 16 + l16] = (_Float16)p;
        atomicAdd(&a_avg[(size_t)(b * T + qr0 + i + hs * 8) * T + key0 +
                         ns * 16 + l16],
                  p * INV_H);
      }
    }

    // rebuild P as A-fragments (K = keys 0..63) from this wave's LDS strip
    FragH ap[2];
#pragma unroll
    for (int f = 0; f < 2; ++f)
#pragma unroll
      for (int v = 0; v < 8; ++v) {
        const int kk = 32 * f + ((v & 3) * 2) + ((v >= 4) ? 16 : 0) + hs * 8;
        ap[f].u[v] = *(const unsigned*)&pbuf[wave][l16][kk];
      }

#pragma unroll
    for (int nd = 0; nd < 4; ++nd) {
#pragma unroll
      for (int f = 0; f < 2; ++f) {
        FragH bv;
#pragma unroll
        for (int v = 0; v < 8; ++v)
          bv.u[v] =
              *(const unsigned*)&vt[nd * 16 + l16][32 * f + ((v & 3) * 2) + ((v >= 4) ? 16 : 0) + hs * 8];
        *oacc[nd] = wmma_f16(ap[f].h, bv.h, *oacc[nd]);
      }
    }
    __syncthreads();
  }

#pragma unroll
  for (int nd = 0; nd < 4; ++nd)
#pragma unroll
    for (int i = 0; i < 8; ++i) {
      const int qrow = qr0 + i + hs * 8;
      ctx[(size_t)(b * T + qrow) * E + h * HD + nd * 16 + l16] =
          (_Float16)((*oacc[nd])[i]);
    }
}

// ---------------------------------------------------------------------------
extern "C" void kernel_launch(void* const* d_in, const int* in_sizes, int n_in,
                              void* d_out, int out_size, void* d_ws,
                              size_t ws_size, hipStream_t stream) {
  const float* x    = (const float*)d_in[0];
  const float* mre  = (const float*)d_in[1];
  const float* mim  = (const float*)d_in[2];
  const float* pad  = (const float*)d_in[3];
  const float* qkvw = (const float*)d_in[4];
  const float* qkvb = (const float*)d_in[5];
  const float* ow   = (const float*)d_in[6];
  const float* ob   = (const float*)d_in[7];

  float* out_o = (float*)d_out;                       // [B,T,E]
  float* a_avg = out_o + (size_t)B * T * E;           // [B,T,T]

  char* w = (char*)d_ws;
  float*    qkvf = (float*)w;                         // [BT][3E] f32, 48 MB
  _Float16* qhp  = (_Float16*)(w + 50331648);         // [B,H,T,HD] f16, 8 MB
  _Float16* khp  = (_Float16*)(w + 58720256);
  _Float16* vhp  = (_Float16*)(w + 67108864);
  _Float16* ctxp = (_Float16*)(w + 75497472);         // [B,T,E] f16, 8 MB

  zero_f32<<<(B * T * T / 4 + 255) / 256, 256, 0, stream>>>(a_avg,
                                                            B * T * T / 4);

  gemm_nt_bias<false><<<dim3(E3 / 64, BT / 64), 128, 0, stream>>>(
      x, qkvw, qkvb, qkvf, BT, E3, E);

  rotary_scatter<<<(B * T * (E3 / 2) + 255) / 256, 256, 0, stream>>>(
      qkvf, mre, mim, qhp, khp, vhp);

  attention<<<B * H * (T / 64), 128, 0, stream>>>(qhp, khp, vhp, pad, a_avg,
                                                  ctxp);

  gemm_nt_bias<true><<<dim3(E / 64, BT / 64), 128, 0, stream>>>(
      ctxp, ow, ob, out_o, BT, E, E);
}
